// GCN_LSTM_67302137528441
// MI455X (gfx1250) — matrix operand order repaired and checked by reading, not compiled, not run on hardware
//
#include <hip/hip_runtime.h>
#include <hip/hip_bf16.h>
#include <math.h>

// Problem constants (from reference): T, N, F_IN, H_G, H_L, C, E
#define TT   12
#define NN   4096
#define FIN  64
#define HG   128
#define HL   128
#define NCLS 10
#define EE   65536

typedef __bf16 v16bf __attribute__((ext_vector_type(16)));
typedef float  v8f   __attribute__((ext_vector_type(8)));

union FragBF { v16bf v; unsigned int u[8]; };

__device__ __forceinline__ unsigned short f2bf(float f) {
    unsigned int u = __float_as_uint(f);
    u += 0x7FFFu + ((u >> 16) & 1u);      // round-to-nearest-even
    return (unsigned short)(u >> 16);
}
__device__ __forceinline__ float sigm(float x) { return 1.0f / (1.0f + expf(-x)); }

// ---------------------------------------------------------------------------
// Small utility kernels
// ---------------------------------------------------------------------------
__global__ void zero_f32(float* __restrict__ p, int n) {
    int i = blockIdx.x * blockDim.x + threadIdx.x;
    if (i < n) p[i] = 0.0f;
}
__global__ void fill_f32(float* __restrict__ p, float v, int n) {
    int i = blockIdx.x * blockDim.x + threadIdx.x;
    if (i < n) p[i] = v;
}
__global__ void f32_to_bf16(const float* __restrict__ in, unsigned short* __restrict__ out, int n) {
    int i = blockIdx.x * blockDim.x + threadIdx.x;
    if (i < n) out[i] = f2bf(in[i]);
}
// in[rows][cols] (f32) -> out[cols][rows] (bf16)
__global__ void transpose_to_bf16(const float* __restrict__ in, unsigned short* __restrict__ out,
                                  int rows, int cols) {
    int i = blockIdx.x * blockDim.x + threadIdx.x;
    if (i >= rows * cols) return;
    int r = i / cols, c = i - r * cols;
    out[c * rows + r] = f2bf(in[i]);
}
// Wfc[10][128] -> WfcT bf16 [128][16], columns >= 10 zero-padded
__global__ void build_wfcT(const float* __restrict__ wfc, unsigned short* __restrict__ out) {
    int i = blockIdx.x * blockDim.x + threadIdx.x;  // < 128*16
    if (i >= 128 * 16) return;
    int k = i >> 4, n = i & 15;
    out[i] = (n < NCLS) ? f2bf(wfc[n * 128 + k]) : (unsigned short)0;
}
__global__ void add_bias_vec(const float* __restrict__ a, const float* __restrict__ b,
                             float* __restrict__ o, int n) {
    int i = blockIdx.x * blockDim.x + threadIdx.x;
    if (i < n) o[i] = a[i] + b[i];
}

// ---------------------------------------------------------------------------
// Graph normalization
// ---------------------------------------------------------------------------
__global__ void deg_accum(const int* __restrict__ col, const float* __restrict__ w,
                          float* __restrict__ deg) {
    int e = blockIdx.x * blockDim.x + threadIdx.x;
    if (e < EE) atomicAdd(&deg[col[e]], w[e]);
}
__global__ void make_dinv(const float* __restrict__ deg, float* __restrict__ dinv) {
    int n = blockIdx.x * blockDim.x + threadIdx.x;
    if (n < NN) dinv[n] = (deg[n] > 0.0f) ? rsqrtf(deg[n]) : 0.0f;
}
__global__ void make_norm(const int* __restrict__ row, const int* __restrict__ col,
                          const float* __restrict__ w, const float* __restrict__ dinv,
                          float* __restrict__ norm) {
    int e = blockIdx.x * blockDim.x + threadIdx.x;
    if (e < EE) norm[e] = dinv[row[e]] * w[e] * dinv[col[e]];
}

// ---------------------------------------------------------------------------
// bf16 WMMA GEMM: D[M x Nout] = A[M x K] * B[K x Nout], f32 accumulate.
// One wave per 16 x (NT*16) output strip: A fragment loaded once per K-step
// and reused across NT WMMAs (A traffic cut NT-fold vs one-tile-per-wave).
// Layouts per CDNA5 ISA 7.12.2/7.12.4 (A: lane = M row + K-half; B: lane = N col + K-half,
// vgpr = K pair; C/D: vgpr r -> M = r + 8*(lane>=16), N = lane&15).
// ---------------------------------------------------------------------------
template <int NT>
__global__ void __launch_bounds__(32)
gemm_bf16_wmma(const unsigned int* __restrict__ A32,   // [M][K/2] dwords
               const unsigned int* __restrict__ B32,   // [K][Nout/2] dwords
               float* __restrict__ D, int M, int K, int Nout) {
    const int lane = threadIdx.x & 31;
    const int m    = lane & 15;
    const int half = lane >> 4;
    const int mt   = blockIdx.x;
    const int ng   = blockIdx.y;                       // group of NT n-tiles
    const int K2   = K >> 1;
    const int N2   = Nout >> 1;
    const unsigned int* arow = A32 + (size_t)(mt * 16 + m) * K2;
    const int nb2 = ng * NT * 8;                       // dword offset of strip

    v8f acc[NT] = {};
    for (int kb = 0; kb < K; kb += 32) {
        FragBF a;
#pragma unroll
        for (int v = 0; v < 8; v++) {
            int k0 = kb + ((v >> 2) << 4) + (half << 3) + ((v & 3) << 1);
            a.u[v] = arow[k0 >> 1];
        }
        const unsigned short* B16 = (const unsigned short*)B32;
#pragma unroll
        for (int nt = 0; nt < NT; nt++) {
            FragBF b;
            const int bcol = ng * NT * 16 + nt * 16 + m;
#pragma unroll
            for (int v = 0; v < 8; v++) {
                int k0 = kb + ((v >> 2) << 4) + (half << 3) + ((v & 3) << 1);
                b.u[v] = (unsigned int)B16[(size_t)k0 * Nout + bcol] | ((unsigned int)B16[(size_t)(k0 + 1) * Nout + bcol] << 16);
            }
            acc[nt] = __builtin_amdgcn_wmma_f32_16x16x32_bf16(false, a.v, false, b.v,
                                                              (short)0, acc[nt], false, false);
        }
    }
#pragma unroll
    for (int nt = 0; nt < NT; nt++)
#pragma unroll
        for (int r = 0; r < 8; r++)
            D[(size_t)(mt * 16 + r + (half << 3)) * Nout + ng * NT * 16 + nt * 16 + m] = acc[nt][r];
}

// ---------------------------------------------------------------------------
// GCN edge aggregation: agg[t, col[e], h] += norm[e] * hw[t, row[e], h]
// ---------------------------------------------------------------------------
__global__ void __launch_bounds__(128)
gcn_aggregate(const int* __restrict__ row, const int* __restrict__ col,
              const float* __restrict__ norm, const float* __restrict__ hw,
              float* __restrict__ agg) {
    const int e = blockIdx.x;
    const int h = threadIdx.x;
    const int s = row[e], d = col[e];
    const float wn = norm[e];
#pragma unroll
    for (int t = 0; t < TT; t++) {
        atomicAdd(&agg[((size_t)t * NN + d) * HG + h],
                  wn * hw[((size_t)t * NN + s) * HG + h]);
    }
}

// agg + selfnorm*hw + bias -> exact GELU -> bf16 (normal [t][n][h] or scan [n][t][h])
__global__ void gcn_bias_gelu(const float* __restrict__ agg, const float* __restrict__ hw,
                              const float* __restrict__ dinv, const float* __restrict__ bias,
                              unsigned short* __restrict__ out_bf, int scan) {
    int idx = blockIdx.x * blockDim.x + threadIdx.x;   // < TT*NN*HG
    int h = idx & 127;
    int n = (idx >> 7) & (NN - 1);
    int t = idx >> 19;                                  // / (NN*HG) = / 2^19
    float sn = dinv[n] * dinv[n];
    float v = agg[idx] + sn * hw[idx] + bias[h];
    float g = 0.5f * v * (1.0f + erff(v * 0.70710678118f));
    int dst = scan ? ((n * TT + t) * HG + h) : idx;
    out_bf[dst] = f2bf(g);
}

// ---------------------------------------------------------------------------
// Persistent LSTM scan (reference scans over N=4096 steps, batch B=12).
// Single workgroup, 256 threads = 8 waves; Whh^T (bf16, 128KB) LDS-resident.
// HASGX: input projections precomputed (Gx = XS @ Wih^T, all steps, parallel
//        WMMA GEMM); per step only the recurrent H @ Whh^T chain runs
//        (4 dependent WMMAs per acc instead of 8 -> half the critical path).
// !HASGX fallback (workspace too small for Gx): Wih^T also LDS-resident and
//        X @ Wih^T computed in-loop.
// ---------------------------------------------------------------------------
#define LSTM_SMEM_BASE (128*256*4 /*Whh*/ + 16*512*4 /*G*/ + 16*64*4 /*H*/ + 12*128*4 /*C*/)
#define LSTM_SMEM_FUSED (LSTM_SMEM_BASE + 128*256*4 /*Wih*/)

template <bool HASGX, bool WHS, bool WLAST>
__global__ void __launch_bounds__(256)
lstm_scan_kernel(const unsigned int* __restrict__ xs32,    // [nsteps][12][64] dwords (bf16) (!HASGX)
                 const float* __restrict__ Gx,             // [nsteps][12][512] f32 (HASGX)
                 const unsigned int* __restrict__ WihT32,  // [128][256] dwords (!HASGX)
                 const unsigned int* __restrict__ WhhT32,  // [128][256] dwords
                 const float* __restrict__ cb,             // [512] combined bias
                 unsigned short* __restrict__ hs_out,      // [nsteps][12][128] bf16 (WHS)
                 float* __restrict__ last_out,             // [nsteps][128] (WLAST, b==11)
                 int nsteps) {
    extern __shared__ char smem[];
    unsigned int*   sWhh = (unsigned int*)smem;                 // 32768 dwords
    float*          sG   = (float*)(sWhh + 128 * 256);          // 16*512 f32
    unsigned int*   sH32 = (unsigned int*)(sG + 16 * 512);      // 16*128 bf16 as dwords
    unsigned short* sH16 = (unsigned short*)sH32;
    float*          sC   = (float*)(sH32 + 16 * 64);            // 12*128 f32
    unsigned int*   sWih = (unsigned int*)(sC + 12 * 128);      // 32768 dwords (!HASGX only)

    const int tid = threadIdx.x;
    for (int i = tid; i < 128 * 256; i += 256) sWhh[i] = WhhT32[i];
    if constexpr (!HASGX)
        for (int i = tid; i < 128 * 256; i += 256) sWih[i] = WihT32[i];
    for (int i = tid; i < 16 * 64; i += 256) sH32[i] = 0u;      // h rows 12..15 stay zero
    for (int i = tid; i < 12 * 128; i += 256) sC[i] = 0.0f;
    __syncthreads();

    const int w = tid >> 5, lane = tid & 31;
    const int m = lane & 15, half = lane >> 4;
    const int nbase = w << 6;                                   // this wave's 64 gate columns

    for (int n = 0; n < nsteps; n++) {
        v8f acc[4];
        if constexpr (HASGX) {
            // Seed accumulators with precomputed input projections.
            const float* gx = Gx + (size_t)n * 12 * 512;
            if (n + 1 < nsteps)
                __builtin_prefetch(Gx + (size_t)(n + 1) * 12 * 512 + tid * 24, 0, 0);
#pragma unroll
            for (int nt = 0; nt < 4; nt++) {
                const int colj = nbase + (nt << 4) + m;
#pragma unroll
                for (int r = 0; r < 8; r++) {
                    int rr = r + (half << 3);
                    acc[nt][r] = gx[(rr < 12 ? rr : 11) * 512 + colj]; // rows 12..15 dummy
                }
            }
        } else {
            const unsigned int* xrow = xs32 + (size_t)n * 12 * 64;
            if (n + 1 < nsteps)
                __builtin_prefetch(xs32 + (size_t)(n + 1) * 12 * 64 + tid * 3, 0, 0);
#pragma unroll
            for (int nt = 0; nt < 4; nt++) acc[nt] = (v8f){};
            // G += X @ Wih^T   (M padded 12->16, garbage rows masked to zero)
#pragma unroll
            for (int q = 0; q < 4; q++) {
                const int kb = q << 5;
                FragBF a;
#pragma unroll
                for (int v = 0; v < 8; v++) {
                    int k0 = kb + ((v >> 2) << 4) + (half << 3) + ((v & 3) << 1);
                    a.u[v] = (m < 12) ? xrow[m * 64 + (k0 >> 1)] : 0u;
                }
                const unsigned short* sWih16 = (const unsigned short*)sWih;
#pragma unroll
                for (int nt = 0; nt < 4; nt++) {
                    FragBF b;
                    const int nb2 = (nbase + (nt << 4)) >> 1;
#pragma unroll
                    for (int v = 0; v < 8; v++) {
                        int k0 = kb + ((v >> 2) << 4) + (half << 3) + ((v & 3) << 1);
                        int bc = nbase + (nt << 4) + m;
                        b.u[v] = (unsigned int)sWih16[k0 * 512 + bc] | ((unsigned int)sWih16[(k0 + 1) * 512 + bc] << 16);
                    }
                    acc[nt] = __builtin_amdgcn_wmma_f32_16x16x32_bf16(
                        false, a.v, false, b.v, (short)0, acc[nt], false, false);
                }
            }
        }
        // G += H @ Whh^T  (the recurrent critical path)
#pragma unroll
        for (int q = 0; q < 4; q++) {
            const int kb = q << 5;
            FragBF a;
#pragma unroll
            for (int v = 0; v < 8; v++) {
                int k0 = kb + ((v >> 2) << 4) + (half << 3) + ((v & 3) << 1);
                a.u[v] = sH32[m * 64 + (k0 >> 1)];
            }
            const unsigned short* sWhh16 = (const unsigned short*)sWhh;
#pragma unroll
            for (int nt = 0; nt < 4; nt++) {
                FragBF b;
                const int nb2 = (nbase + (nt << 4)) >> 1;
#pragma unroll
                for (int v = 0; v < 8; v++) {
                    int k0 = kb + ((v >> 2) << 4) + (half << 3) + ((v & 3) << 1);
                    int bc = nbase + (nt << 4) + m;
                    b.u[v] = (unsigned int)sWhh16[k0 * 512 + bc] | ((unsigned int)sWhh16[(k0 + 1) * 512 + bc] << 16);
                }
                acc[nt] = __builtin_amdgcn_wmma_f32_16x16x32_bf16(
                    false, a.v, false, b.v, (short)0, acc[nt], false, false);
            }
        }
        // Stage gate pre-activations to LDS
#pragma unroll
        for (int nt = 0; nt < 4; nt++) {
            const int colj = nbase + (nt << 4) + m;
#pragma unroll
            for (int r = 0; r < 8; r++)
                sG[(r + (half << 3)) * 512 + colj] = acc[nt][r];
        }
        __syncthreads();

        // Gate nonlinearities + state update (b<12 only)
        for (int i = tid; i < 12 * 128; i += 256) {
            const int b = i >> 7, j = i & 127;
            const float gi = sG[b * 512 + j]       + cb[j];
            const float gf = sG[b * 512 + 128 + j] + cb[128 + j];
            const float gg = sG[b * 512 + 256 + j] + cb[256 + j];
            const float go = sG[b * 512 + 384 + j] + cb[384 + j];
            const float c  = sigm(gf) * sC[i] + sigm(gi) * tanhf(gg);
            const float h  = sigm(go) * tanhf(c);
            sC[i] = c;
            sH16[b * 128 + j] = f2bf(h);
            if constexpr (WHS)   hs_out[((size_t)n * 12 + b) * 128 + j] = f2bf(h);
            if constexpr (WLAST) if (b == 11) last_out[(size_t)n * 128 + j] = h;
        }
        __syncthreads();
    }
}

// out[n][c] = fcl[n][c(<16)] + bfc[c]
__global__ void fc_final(const float* __restrict__ fcl, const float* __restrict__ bfc,
                         float* __restrict__ out) {
    int i = blockIdx.x * blockDim.x + threadIdx.x;   // < NN*NCLS
    if (i >= NN * NCLS) return;
    int n = i / NCLS, c = i - n * NCLS;
    out[i] = fcl[n * 16 + c] + bfc[c];
}

// ---------------------------------------------------------------------------
// Host side
// ---------------------------------------------------------------------------
static inline size_t alignup(size_t x) { return (x + 255) & ~(size_t)255; }

extern "C" void kernel_launch(void* const* d_in, const int* in_sizes, int n_in,
                              void* d_out, int out_size, void* d_ws, size_t ws_size,
                              hipStream_t stream) {
    (void)in_sizes; (void)n_in; (void)out_size;

    const float* x    = (const float*)d_in[0];
    const int*   ei   = (const int*)d_in[1];
    const float* ew   = (const float*)d_in[2];
    const float* W1   = (const float*)d_in[3];
    const float* b1   = (const float*)d_in[4];
    const float* W2   = (const float*)d_in[5];
    const float* b2   = (const float*)d_in[6];
    const float* W3   = (const float*)d_in[7];
    const float* b3   = (const float*)d_in[8];
    const float* Wih1 = (const float*)d_in[9];
    const float* Whh1 = (const float*)d_in[10];
    const float* bih1 = (const float*)d_in[11];
    const float* bhh1 = (const float*)d_in[12];
    const float* Wih2 = (const float*)d_in[13];
    const float* Whh2 = (const float*)d_in[14];
    const float* bih2 = (const float*)d_in[15];
    const float* bhh2 = (const float*)d_in[16];
    const float* Wfc  = (const float*)d_in[17];
    const float* bfc  = (const float*)d_in[18];
    const int* erow = ei;
    const int* ecol = ei + EE;
    float* out = (float*)d_out;

    // Workspace carve-up (~93 MB base, +100 MB optional Gx buffer)
    char* base = (char*)d_ws;
    size_t off = 0;
    auto carve = [&](size_t bytes) { char* p = base + off; off += alignup(bytes); return p; };
    float*          deg     = (float*)carve(NN * 4);
    float*          dinv    = (float*)carve(NN * 4);
    float*          normv   = (float*)carve((size_t)EE * 4);
    unsigned short* W1b     = (unsigned short*)carve(FIN * HG * 2);
    unsigned short* W2b     = (unsigned short*)carve(HG * HG * 2);
    unsigned short* W3b     = (unsigned short*)carve(HG * HG * 2);
    unsigned short* WihT1b  = (unsigned short*)carve(128 * 512 * 2);
    unsigned short* WhhT1b  = (unsigned short*)carve(128 * 512 * 2);
    unsigned short* WihT2b  = (unsigned short*)carve(128 * 512 * 2);
    unsigned short* WhhT2b  = (unsigned short*)carve(128 * 512 * 2);
    float*          cb1     = (float*)carve(512 * 4);
    float*          cb2     = (float*)carve(512 * 4);
    unsigned short* WfcTb   = (unsigned short*)carve(128 * 16 * 2);
    unsigned short* xb      = (unsigned short*)carve((size_t)TT * NN * HG * 2); // GEMM A (bf16)
    float*          hw      = (float*)carve((size_t)TT * NN * HG * 4);
    float*          agg     = (float*)carve((size_t)TT * NN * HG * 4);
    unsigned short* xs_scan = (unsigned short*)carve((size_t)NN * TT * HG * 2); // GCN3 out, scan layout
    unsigned short* hs1     = (unsigned short*)carve((size_t)NN * TT * HG * 2); // LSTM-1 out, scan layout
    float*          lastf   = (float*)carve((size_t)NN * HL * 4);
    unsigned short* lastb   = (unsigned short*)carve((size_t)NN * HL * 2);
    float*          fcl     = (float*)carve((size_t)NN * 16 * 4);
    const size_t gx_bytes = (size_t)NN * TT * 512 * 4;          // 100.7 MB
    const bool   use_gx   = (off + alignup(gx_bytes)) <= ws_size;
    float*          Gx      = use_gx ? (float*)carve(gx_bytes) : nullptr;

    const int BT  = 256;
    const int TNH = TT * NN * HG;                  // 6291456

    // ---- Graph normalization -------------------------------------------------
    fill_f32<<<(NN + BT - 1) / BT, BT, 0, stream>>>(deg, 1.0f, NN);   // self-loop weight 1
    deg_accum<<<(EE + BT - 1) / BT, BT, 0, stream>>>(ecol, ew, deg);
    make_dinv<<<(NN + BT - 1) / BT, BT, 0, stream>>>(deg, dinv);
    make_norm<<<(EE + BT - 1) / BT, BT, 0, stream>>>(erow, ecol, ew, dinv, normv);

    // ---- Weight / input conversions -----------------------------------------
    f32_to_bf16<<<(TT * NN * FIN + BT - 1) / BT, BT, 0, stream>>>(x, xb, TT * NN * FIN);
    f32_to_bf16<<<(FIN * HG + BT - 1) / BT, BT, 0, stream>>>(W1, W1b, FIN * HG);
    f32_to_bf16<<<(HG * HG + BT - 1) / BT, BT, 0, stream>>>(W2, W2b, HG * HG);
    f32_to_bf16<<<(HG * HG + BT - 1) / BT, BT, 0, stream>>>(W3, W3b, HG * HG);
    transpose_to_bf16<<<(512 * 128 + BT - 1) / BT, BT, 0, stream>>>(Wih1, WihT1b, 512, 128);
    transpose_to_bf16<<<(512 * 128 + BT - 1) / BT, BT, 0, stream>>>(Whh1, WhhT1b, 512, 128);
    transpose_to_bf16<<<(512 * 128 + BT - 1) / BT, BT, 0, stream>>>(Wih2, WihT2b, 512, 128);
    transpose_to_bf16<<<(512 * 128 + BT - 1) / BT, BT, 0, stream>>>(Whh2, WhhT2b, 512, 128);
    add_bias_vec<<<2, BT, 0, stream>>>(bih1, bhh1, cb1, 512);
    add_bias_vec<<<2, BT, 0, stream>>>(bih2, bhh2, cb2, 512);
    build_wfcT<<<(128 * 16 + BT - 1) / BT, BT, 0, stream>>>(Wfc, WfcTb);

    // ---- 3 GCN layers --------------------------------------------------------
    const unsigned short* Ws[3] = { W1b, W2b, W3b };
    const float*          bs[3] = { b1, b2, b3 };
    const int             Ks[3] = { FIN, HG, HG };
    for (int layer = 0; layer < 3; layer++) {
        gemm_bf16_wmma<8><<<dim3(TT * NN / 16, 1), 32, 0, stream>>>(
            (const unsigned int*)xb, (const unsigned int*)Ws[layer], hw,
            TT * NN, Ks[layer], HG);
        zero_f32<<<(TNH + BT - 1) / BT, BT, 0, stream>>>(agg, TNH);
        gcn_aggregate<<<EE, 128, 0, stream>>>(erow, ecol, normv, hw, agg);
        gcn_bias_gelu<<<TNH / BT, BT, 0, stream>>>(agg, hw, dinv, bs[layer],
                                                   (layer == 2) ? xs_scan : xb,
                                                   (layer == 2) ? 1 : 0);
    }

    // ---- 2-layer LSTM scan over N=4096 steps (batch 12) ---------------------
    if (use_gx) {
        // Precompute input projections for ALL steps as parallel WMMA GEMMs;
        // the sequential scan then only carries H @ Whh^T.
        gemm_bf16_wmma<8><<<dim3(TT * NN / 16, 4), 32, 0, stream>>>(
            (const unsigned int*)xs_scan, (const unsigned int*)WihT1b, Gx,
            TT * NN, HL, 512);
        lstm_scan_kernel<true, true, false><<<1, 256, LSTM_SMEM_BASE, stream>>>(
            nullptr, Gx, nullptr, (const unsigned int*)WhhT1b, cb1, hs1, nullptr, NN);
        gemm_bf16_wmma<8><<<dim3(TT * NN / 16, 4), 32, 0, stream>>>(
            (const unsigned int*)hs1, (const unsigned int*)WihT2b, Gx,
            TT * NN, HL, 512);
        lstm_scan_kernel<true, false, true><<<1, 256, LSTM_SMEM_BASE, stream>>>(
            nullptr, Gx, nullptr, (const unsigned int*)WhhT2b, cb2, nullptr, lastf, NN);
    } else {
        lstm_scan_kernel<false, true, false><<<1, 256, LSTM_SMEM_FUSED, stream>>>(
            (const unsigned int*)xs_scan, nullptr, (const unsigned int*)WihT1b,
            (const unsigned int*)WhhT1b, cb1, hs1, nullptr, NN);
        lstm_scan_kernel<false, false, true><<<1, 256, LSTM_SMEM_FUSED, stream>>>(
            (const unsigned int*)hs1, nullptr, (const unsigned int*)WihT2b,
            (const unsigned int*)WhhT2b, cb2, nullptr, lastf, NN);
    }

    // ---- FC head -------------------------------------------------------------
    f32_to_bf16<<<(NN * HL + BT - 1) / BT, BT, 0, stream>>>(lastf, lastb, NN * HL);
    gemm_bf16_wmma<1><<<dim3(NN / 16, 1), 32, 0, stream>>>(
        (const unsigned int*)lastb, (const unsigned int*)WfcTb, fcl, NN, HL, 16);
    fc_final<<<(NN * NCLS + BT - 1) / BT, BT, 0, stream>>>(fcl, bfc, out);
}
